// Recall_90975997264408
// MI455X (gfx1250) — compile-verified, matrix operand order checked by reading, not run on hardware
//
#include <hip/hip_runtime.h>
#include <math.h>

typedef _Float16 v4h  __attribute__((ext_vector_type(4)));
typedef _Float16 v8h  __attribute__((ext_vector_type(8)));
typedef _Float16 v16h __attribute__((ext_vector_type(16)));
typedef float    v8f  __attribute__((ext_vector_type(8)));

#define DIM   128
#define LSTR  136   // padded row stride in f16 elems (272B -> bank-conflict-free frags)

// Load one 16x32 f16 fragment (A-layout) from row-major LDS [row][k], stride LSTR.
// Per ISA: lane l -> row (l&15); half h=l>>4; VGPR0-3 hold k = 8h..8h+7, VGPR4-7 hold k = 16+8h..16+8h+7.
// The B-fragment has the identical (lane,reg)->k mapping with the lane index selecting
// the output column, so loading B^T row-major with this same loader is correct.
__device__ __forceinline__ v16h load_frag(const _Float16* base, int row0, int k0, int lane) {
    const int m  = lane & 15;
    const int hf = lane >> 4;
    const _Float16* p = base + (row0 + m) * LSTR + k0 + hf * 8;
    union { v16h v; v8h h8[2]; } u;
    u.h8[0] = *(const v8h*)(p);        // k = k0 + 8h .. +7
    u.h8[1] = *(const v8h*)(p + 16);   // k = k0 + 16 + 8h .. +7
    return u.v;
}

// C = A(16x128 strip at m0) * B  (B given transposed row-major as Bt[128][128]),
// result written to LDS as f16: row-major, or transposed (for building V^T).
__device__ __forceinline__ void gemm_strip(const _Float16* A, const _Float16* Bt,
                                           _Float16* dst, int m0, int lane, bool storeT) {
    v16h a[4];
#pragma unroll
    for (int kk = 0; kk < 4; ++kk) a[kk] = load_frag(A, m0, kk * 32, lane);
    const int n  = lane & 15;
    const int hf = lane >> 4;
#pragma unroll
    for (int t = 0; t < 8; ++t) {
        v8f c = {};
#pragma unroll
        for (int kk = 0; kk < 4; ++kk) {
            v16h bfr = load_frag(Bt, t * 16, kk * 32, lane);
            c = __builtin_amdgcn_wmma_f32_16x16x32_f16(false, a[kk], false, bfr,
                                                       (short)0, c, false, false);
        }
        if (storeT) {
            v8h pk;
#pragma unroll
            for (int r = 0; r < 8; ++r) pk[r] = (_Float16)c[r];
            *(v8h*)(dst + (t * 16 + n) * LSTR + m0 + 8 * hf) = pk;   // dstT[col][row]
        } else {
#pragma unroll
            for (int r = 0; r < 8; ++r)
                dst[(m0 + 8 * hf + r) * LSTR + t * 16 + n] = (_Float16)c[r];
        }
    }
}

__device__ __forceinline__ void load_wT(const float* __restrict__ W, _Float16* Wb, int tid) {
    for (int idx = tid; idx < DIM * DIM; idx += 256) {
        const int i = idx >> 7, j = idx & 127;
        Wb[j * LSTR + i] = (_Float16)W[idx];     // store W^T row-major
    }
}

__global__ __launch_bounds__(256)
void local_attn_kernel(const float* __restrict__ x,  const float* __restrict__ Wq,
                       const float* __restrict__ Wk, const float* __restrict__ Wv,
                       const float* __restrict__ Wo, const float* __restrict__ gamma,
                       const float* __restrict__ beta, float* __restrict__ out) {
    extern __shared__ _Float16 smem[];
    _Float16* Xs = smem;               // X block (f16), later reused for attn@V output
    _Float16* Wb = Xs + DIM * LSTR;    // current weight^T
    _Float16* Qs = Wb + DIM * LSTR;    // Q, later reused for attn probabilities
    _Float16* Ks = Qs + DIM * LSTR;    // K row-major (== (K^T)^T for scores B operand)
    _Float16* Vs = Ks + DIM * LSTR;    // V^T

    const int tid  = threadIdx.x;
    const int lane = tid & 31;
    const int wave = tid >> 5;
    const int m0   = wave * 16;
    const int n    = lane & 15;
    const int hf   = lane >> 4;

    const int wg  = blockIdx.x;        // 512 blocks = 8 batches x 64 windows
    const int b   = wg >> 6;
    const int blk = wg & 63;
    const float* xg = x + ((size_t)b * 8192 + (size_t)blk * 128) * DIM;

    // ---- load X block, f32 -> f16, row-major padded ----
    {
        const float4* xg4 = (const float4*)xg;
        for (int idx = tid; idx < (DIM * DIM) / 4; idx += 256) {
            float4 v = xg4[idx];
            const int row = idx >> 5;
            const int col = (idx & 31) * 4;
            v4h pk;
            pk[0] = (_Float16)v.x; pk[1] = (_Float16)v.y;
            pk[2] = (_Float16)v.z; pk[3] = (_Float16)v.w;
            *(v4h*)(Xs + row * LSTR + col) = pk;
        }
    }
    load_wT(Wq, Wb, tid);
    __syncthreads();

    gemm_strip(Xs, Wb, Qs, m0, lane, false);   // Q
    __syncthreads();
    load_wT(Wk, Wb, tid);
    __syncthreads();
    gemm_strip(Xs, Wb, Ks, m0, lane, false);   // K (row-major)
    __syncthreads();
    load_wT(Wv, Wb, tid);
    __syncthreads();
    gemm_strip(Xs, Wb, Vs, m0, lane, true);    // V^T
    __syncthreads();
    load_wT(Wo, Wb, tid);                      // stage Wo^T for the final GEMM
    __syncthreads();

    // ---- scores = Q K^T / sqrt(D), softmax over the 128 keys ----
    v16h qa[4];
#pragma unroll
    for (int kk = 0; kk < 4; ++kk) qa[kk] = load_frag(Qs, m0, kk * 32, lane);
    v8f sc[8];
#pragma unroll
    for (int t = 0; t < 8; ++t) {
        v8f c = {};
#pragma unroll
        for (int kk = 0; kk < 4; ++kk) {
            v16h bfr = load_frag(Ks, t * 16, kk * 32, lane);
            c = __builtin_amdgcn_wmma_f32_16x16x32_f16(false, qa[kk], false, bfr,
                                                       (short)0, c, false, false);
        }
        sc[t] = c;
    }
    const float scale = 0.0883883476483184f;   // 1/sqrt(128)
#pragma unroll
    for (int r = 0; r < 8; ++r) {
        float mx = -3.0e38f;
#pragma unroll
        for (int t = 0; t < 8; ++t) { float v = sc[t][r] * scale; sc[t][r] = v; mx = fmaxf(mx, v); }
#pragma unroll
        for (int off = 8; off >= 1; off >>= 1) mx = fmaxf(mx, __shfl_xor(mx, off, 32));
        float s = 0.f;
#pragma unroll
        for (int t = 0; t < 8; ++t) { float e = expf(sc[t][r] - mx); sc[t][r] = e; s += e; }
#pragma unroll
        for (int off = 8; off >= 1; off >>= 1) s += __shfl_xor(s, off, 32);
        const float inv = 1.f / s;
#pragma unroll
        for (int t = 0; t < 8; ++t) sc[t][r] *= inv;
    }
    // attn (C-layout) -> row-major f16 into Qs (wave-private rows; Q fully consumed)
#pragma unroll
    for (int t = 0; t < 8; ++t)
#pragma unroll
        for (int r = 0; r < 8; ++r)
            Qs[(m0 + 8 * hf + r) * LSTR + t * 16 + n] = (_Float16)sc[t][r];

    // ---- out = attn @ V  (B operand = V^T row-major) -> Xs (X consumed) ----
    gemm_strip(Qs, Vs, Xs, m0, lane, false);

    // ---- H = out @ Wo, then LayerNorm + exact GELU, store f32 ----
    v16h aa[4];
#pragma unroll
    for (int kk = 0; kk < 4; ++kk) aa[kk] = load_frag(Xs, m0, kk * 32, lane);
    float g[8], be[8];
#pragma unroll
    for (int t = 0; t < 8; ++t) { g[t] = gamma[t * 16 + n]; be[t] = beta[t * 16 + n]; }
    v8f hh[8];
#pragma unroll
    for (int t = 0; t < 8; ++t) {
        v8f c = {};
#pragma unroll
        for (int kk = 0; kk < 4; ++kk) {
            v16h bfr = load_frag(Wb, t * 16, kk * 32, lane);
            c = __builtin_amdgcn_wmma_f32_16x16x32_f16(false, aa[kk], false, bfr,
                                                       (short)0, c, false, false);
        }
        hh[t] = c;
    }
    float* outg = out + (size_t)b * (16384ull * DIM) + (size_t)blk * 128 * DIM;
#pragma unroll
    for (int r = 0; r < 8; ++r) {
        float s = 0.f, q = 0.f;
#pragma unroll
        for (int t = 0; t < 8; ++t) { float v = hh[t][r]; s += v; q += v * v; }
#pragma unroll
        for (int off = 8; off >= 1; off >>= 1) { s += __shfl_xor(s, off, 32); q += __shfl_xor(q, off, 32); }
        const float mu   = s * (1.f / 128.f);
        const float var  = q * (1.f / 128.f) - mu * mu;
        const float rstd = rsqrtf(var + 1e-5f);
        const int row = m0 + 8 * hf + r;
#pragma unroll
        for (int t = 0; t < 8; ++t) {
            float y  = (hh[t][r] - mu) * rstd * g[t] + be[t];
            float ge = 0.5f * y * (1.f + erff(y * 0.70710678118654752f));
            outg[(size_t)row * DIM + t * 16 + n] = ge;
        }
    }
}

// Second half of the sequence is the zero-padded region: h == 0 exactly, so
// y = beta and the output is gelu(beta[d]) broadcast over all rows.
__global__ __launch_bounds__(256)
void tail_kernel(const float* __restrict__ beta, float* __restrict__ out) {
    const unsigned idx = blockIdx.x * 256u + threadIdx.x;   // 8 * 8192 * 128 threads
    const int d = idx & 127;
    const float y  = beta[d];
    const float ge = 0.5f * y * (1.f + erff(y * 0.70710678118654752f));
    const unsigned b   = idx >> 20;        // 8192*128 = 2^20 elements per batch half
    const unsigned rem = idx & 1048575u;
    out[(size_t)b * 2097152ull + 1048576ull + rem] = ge;
}

extern "C" void kernel_launch(void* const* d_in, const int* in_sizes, int n_in,
                              void* d_out, int out_size, void* d_ws, size_t ws_size,
                              hipStream_t stream) {
    const float* x     = (const float*)d_in[0];
    const float* Wq    = (const float*)d_in[1];
    const float* Wk    = (const float*)d_in[2];
    const float* Wv    = (const float*)d_in[3];
    const float* Wo    = (const float*)d_in[4];
    const float* gamma = (const float*)d_in[5];
    const float* beta  = (const float*)d_in[6];
    float* out = (float*)d_out;

    const size_t shmem = (size_t)5 * DIM * LSTR * sizeof(_Float16);  // ~170 KB
    local_attn_kernel<<<dim3(512), dim3(256), shmem, stream>>>(x, Wq, Wk, Wv, Wo,
                                                               gamma, beta, out);
    tail_kernel<<<dim3(32768), dim3(256), 0, stream>>>(beta, out);
}
